// GraphTransformer_28819230556667
// MI455X (gfx1250) — matrix operand order repaired and checked by reading, not compiled, not run on hardware
//
#include <hip/hip_runtime.h>
#include <hip/hip_bf16.h>

// ---------------- common helpers ----------------

typedef __attribute__((ext_vector_type(16))) __bf16 v16bf;
typedef __attribute__((ext_vector_type(8)))  float  v8f;

union Frag16 {            // 16 bf16 == 8 dwords
    v16bf v;
    unsigned int u[8];
};

__device__ inline unsigned short f32_to_bf16(float f) {
    unsigned int u = __float_as_uint(f);
    unsigned int r = (u + 0x7FFFu + ((u >> 16) & 1u)) >> 16;   // RNE
    return (unsigned short)r;
}

// order-preserving float->uint key (for atomic segment-max)
__device__ inline unsigned int fkey(float f) {
    unsigned int u = __float_as_uint(f);
    return (u & 0x80000000u) ? ~u : (u | 0x80000000u);
}
__device__ inline float funkey(unsigned int k) {
    unsigned int u = (k & 0x80000000u) ? (k ^ 0x80000000u) : ~k;
    return __uint_as_float(u);
}

// ---- gfx1250 async global->LDS copy (16B per lane), with safe fallback ----
#if __has_builtin(__builtin_amdgcn_global_load_async_to_lds_b128) && \
    __has_builtin(__builtin_amdgcn_s_wait_asynccnt)
#define GT_ASYNC_LDS 1
#else
#define GT_ASYNC_LDS 0
#endif

typedef int gt_v4i __attribute__((vector_size(4 * sizeof(int))));
typedef __attribute__((address_space(1))) gt_v4i* gt_gptr;
typedef __attribute__((address_space(3))) gt_v4i* gt_lptr;

__device__ inline void gt_cp16(const unsigned short* g, unsigned short* l) {
#if GT_ASYNC_LDS
    __builtin_amdgcn_global_load_async_to_lds_b128(
        (gt_gptr)(__attribute__((address_space(1))) void*)g,
        (gt_lptr)(__attribute__((address_space(3))) void*)l, 0, 0);
#else
    *reinterpret_cast<uint4*>(l) = *reinterpret_cast<const uint4*>(g);
#endif
}

// ---------------- elementwise / setup kernels ----------------

__global__ void gt_cvt_bf16(const float* __restrict__ in, unsigned short* __restrict__ out, long n) {
    long t = (long)blockIdx.x * blockDim.x + threadIdx.x;
    if (t < n) out[t] = f32_to_bf16(in[t]);
}

__global__ void gt_fill_u32(unsigned int* __restrict__ p, unsigned int v, long n) {
    long t = (long)blockIdx.x * blockDim.x + threadIdx.x;
    if (t < n) p[t] = v;
}

// fuse 4 [K,128] weights into one bf16 TRANSPOSED matrix Bt[512][K] (rows: q|k|v|s cols)
// K is a power of two; kshift = log2(K)
__global__ void gt_concat4_wT(const float* __restrict__ a, const float* __restrict__ b,
                              const float* __restrict__ c, const float* __restrict__ d,
                              unsigned short* __restrict__ out, int K, int kshift) {
    int t = blockIdx.x * blockDim.x + threadIdx.x;
    if (t >= 512 * K) return;
    int col = t >> kshift;          // 0..511  (output channel)
    int k   = t & (K - 1);          // 0..K-1  (input channel)
    int sel = col >> 7, cc = col & 127;
    const float* W = (sel == 0) ? a : (sel == 1) ? b : (sel == 2) ? c : d;
    out[t] = f32_to_bf16(W[k * 128 + cc]);   // out[col*K + k]
}

__global__ void gt_concat4_b(const float* __restrict__ a, const float* __restrict__ b,
                             const float* __restrict__ c, const float* __restrict__ d,
                             float* __restrict__ out) {
    int t = threadIdx.x;               // 512 threads
    int sel = t >> 7, cc = t & 127;
    out[t] = ((sel == 0) ? a : (sel == 1) ? b : (sel == 2) ? c : d)[cc];
}

// classifier weight [128,16] -> transposed + zero-padded bf16 [64][128]
__global__ void gt_cvt_wlT(const float* __restrict__ Wl, unsigned short* __restrict__ out) {
    int t = blockIdx.x * blockDim.x + threadIdx.x;
    if (t >= 64 * 128) return;
    int col = t >> 7, k = t & 127;
    out[t] = (col < 16) ? f32_to_bf16(Wl[k * 16 + col]) : (unsigned short)0;
}

// ---------------- WMMA bf16 GEMM ----------------
//  C[M,Ncols] = A[M,K](bf16, row-major) * Bt[NcolsPad,K](bf16, B transposed) + bias
//  block = 256 threads (8 waves). Tile: 128 rows x 64 cols. K multiple of 32.
//  Double-buffered LDS, async global->LDS staging pipelined one k-step ahead.

#define TILE_M 128
#define TILE_N 64
#define LDSTR  40   // bf16 elements per LDS row (32 + 8 pad) -> conflict-free fragment reads

__global__ __launch_bounds__(256) void gt_gemm_bf16(
    const unsigned short* __restrict__ A, int lda,
    const unsigned short* __restrict__ Bt,
    const float* __restrict__ bias,
    float* __restrict__ C, int ldc,
    int M, int K, int Ncols)
{
    __shared__ unsigned short As[2][TILE_M * LDSTR];
    __shared__ unsigned short Bs[2][TILE_N * LDSTR];

    const int tid   = threadIdx.x;
    const int lane  = tid & 31;
    const int wave  = tid >> 5;                 // 0..7
    const int mBase = blockIdx.x * TILE_M;
    const int nBase = blockIdx.y * TILE_N;

    // staging coordinates: A tile = 512 x 16B chunks (2/thread), B tile = 256 x 16B (1/thread)
    const int ar0 = tid >> 2;                   // rows 0..63
    const int ar1 = ar0 + 64;                   // rows 64..127
    const int akk = (tid & 3) * 8;              // k offset {0,8,16,24}
    const int ga0 = min(mBase + ar0, M - 1);    // clamp instead of branch
    const int ga1 = min(mBase + ar1, M - 1);
    const int bcc = tid >> 2;                   // col 0..63
    const int bkk = (tid & 3) * 8;

    v8f acc[TILE_N / 16];
    const v8f zacc = {0.f, 0.f, 0.f, 0.f, 0.f, 0.f, 0.f, 0.f};
#pragma unroll
    for (int i = 0; i < TILE_N / 16; ++i) acc[i] = zacc;

    auto stage = [&](int buf, int k0) {
        gt_cp16(A + (size_t)ga0 * lda + k0 + akk, &As[buf][ar0 * LDSTR + akk]);
        gt_cp16(A + (size_t)ga1 * lda + k0 + akk, &As[buf][ar1 * LDSTR + akk]);
        gt_cp16(Bt + (size_t)(nBase + bcc) * K + k0 + bkk, &Bs[buf][bcc * LDSTR + bkk]);
    };

    const int nsteps = K >> 5;
    stage(0, 0);

    for (int s = 0; s < nsteps; ++s) {
        const int buf = s & 1;
        if (s + 1 < nsteps) {
            stage(buf ^ 1, (s + 1) << 5);       // prefetch next tile into other buffer
#if GT_ASYNC_LDS
            __builtin_amdgcn_s_wait_asynccnt(3);  // current tile's 3 copies complete
#endif
        } else {
#if GT_ASYNC_LDS
            __builtin_amdgcn_s_wait_asynccnt(0);
#endif
        }
        __syncthreads();

        // --- A fragment (16x32 bf16, ISA lane layout) ---
        Frag16 a;
        {
            const int g = lane >> 4;
            const int r = wave * 16 + (lane & 15);
            const unsigned short* as = &As[buf][r * LDSTR];
#pragma unroll
            for (int j = 0; j < 4; ++j) {
                a.u[j]     = *reinterpret_cast<const unsigned int*>(as + g * 8 + 2 * j);
                a.u[4 + j] = *reinterpret_cast<const unsigned int*>(as + 16 + g * 8 + 2 * j);
            }
        }
        // --- all 4 B fragments first, then back-to-back WMMAs ---
        Frag16 b[TILE_N / 16];
#pragma unroll
        for (int nb = 0; nb < TILE_N / 16; ++nb) {
            const unsigned short* bs =
                &Bs[buf][(nb * 16 + (lane & 15)) * LDSTR + (lane >> 4) * 8];
#pragma unroll
            for (int j = 0; j < 8; ++j)
                b[nb].u[j] = *reinterpret_cast<const unsigned int*>(bs + 2 * j + (j & 4) * 2);
        }
#pragma unroll
        for (int nb = 0; nb < TILE_N / 16; ++nb)
            acc[nb] = __builtin_amdgcn_wmma_f32_16x16x32_bf16(
                false, a.v, false, b[nb].v, (short)0, acc[nb], false, false);

        __syncthreads();
    }

    // --- epilogue: D layout = VGPR i -> row (lane>>4)*8+i, col lane&15 ---
    const int rowInTile = wave * 16 + (lane >> 4) * 8;
    const int col0      = lane & 15;
#pragma unroll
    for (int nb = 0; nb < TILE_N / 16; ++nb) {
        int gc = nBase + nb * 16 + col0;
        if (gc >= Ncols) continue;
        float bval = bias[gc];
#pragma unroll
        for (int i = 0; i < 8; ++i) {
            int gr = mBase + rowInTile + i;
            if (gr < M) C[(size_t)gr * ldc + gc] = acc[nb][i] + bval;
        }
    }
}

// ---------------- edge pass 1: logits + segment max ----------------
// qkvs layout per node: [q(128) | k(128) | v(128) | s(128)]

__global__ void gt_edge_logits(const float* __restrict__ qkvs,
                               const int* __restrict__ src, const int* __restrict__ dst,
                               float* __restrict__ alpha, unsigned int* __restrict__ mkey, int E)
{
    int t = blockIdx.x * blockDim.x + threadIdx.x;
    if (t >= E * 4) return;
    int e = t >> 2, h = t & 3;
    int s = src[e], d = dst[e];
    const float4* qp = reinterpret_cast<const float4*>(qkvs + (size_t)d * 512 + h * 32);
    const float4* kp = reinterpret_cast<const float4*>(qkvs + (size_t)s * 512 + 128 + h * 32);
    float sum = 0.f;
#pragma unroll
    for (int i = 0; i < 8; ++i) {
        float4 qv = qp[i], kv = kp[i];
        sum += qv.x * kv.x + qv.y * kv.y + qv.z * kv.z + qv.w * kv.w;
    }
    float a = sum * 0.17677669529663687f;      // 1/sqrt(32)
    alpha[t] = a;
    atomicMax(&mkey[d * 4 + h], fkey(a));
}

// ---------------- edge pass 2: exp, denom, scatter ea*v[src] -> out[dst] ----------------
// one 32-lane wave per edge; lane -> (head = lane>>3, 4 channels = (lane&7)*4)

__global__ void gt_edge_scatter(const float* __restrict__ qkvs, const float* __restrict__ alpha,
                                const unsigned int* __restrict__ mkey,
                                const int* __restrict__ src, const int* __restrict__ dst,
                                float* __restrict__ denom, float* __restrict__ outacc, int E)
{
    int gt = blockIdx.x * blockDim.x + threadIdx.x;
    int e  = gt >> 5;
    if (e >= E) return;
    int lane = gt & 31;
    int s = src[e], d = dst[e];
    int h  = lane >> 3;
    int c4 = (lane & 7) * 4;
    float m  = funkey(mkey[d * 4 + h]);
    float ea = __expf(alpha[e * 4 + h] - m);
    if ((lane & 7) == 0) unsafeAtomicAdd(&denom[d * 4 + h], ea);
    float4 v = *reinterpret_cast<const float4*>(qkvs + (size_t)s * 512 + 256 + h * 32 + c4);
    float* op = outacc + (size_t)d * 128 + h * 32 + c4;
    unsafeAtomicAdd(op + 0, ea * v.x);
    unsafeAtomicAdd(op + 1, ea * v.y);
    unsafeAtomicAdd(op + 2, ea * v.z);
    unsafeAtomicAdd(op + 3, ea * v.w);
}

// ---------------- node epilogue: normalize + skip + relu -> bf16 ----------------

__global__ void gt_node_epilogue(const float* __restrict__ qkvs, const float* __restrict__ outacc,
                                 const float* __restrict__ denom,
                                 unsigned short* __restrict__ hb, int N)
{
    int t = blockIdx.x * blockDim.x + threadIdx.x;
    if (t >= N * 128) return;
    int n = t >> 7, j = t & 127, h = j >> 5;
    float val = outacc[t] / (denom[n * 4 + h] + 1e-16f) + qkvs[(size_t)n * 512 + 384 + j];
    val = fmaxf(val, 0.f);
    hb[t] = f32_to_bf16(val);
}

// ---------------- host-side orchestration ----------------

extern "C" void kernel_launch(void* const* d_in, const int* in_sizes, int n_in,
                              void* d_out, int out_size, void* d_ws, size_t ws_size,
                              hipStream_t stream) {
    const float* x    = (const float*)d_in[0];
    const int*   edge = (const int*)d_in[1];
    const float* Wq1 = (const float*)d_in[2];  const float* bq1 = (const float*)d_in[3];
    const float* Wk1 = (const float*)d_in[4];  const float* bk1 = (const float*)d_in[5];
    const float* Wv1 = (const float*)d_in[6];  const float* bv1 = (const float*)d_in[7];
    const float* Ws1 = (const float*)d_in[8];  const float* bs1 = (const float*)d_in[9];
    const float* Wq2 = (const float*)d_in[10]; const float* bq2 = (const float*)d_in[11];
    const float* Wk2 = (const float*)d_in[12]; const float* bk2 = (const float*)d_in[13];
    const float* Wv2 = (const float*)d_in[14]; const float* bv2 = (const float*)d_in[15];
    const float* Ws2 = (const float*)d_in[16]; const float* bs2 = (const float*)d_in[17];
    const float* Wl  = (const float*)d_in[18]; const float* bl  = (const float*)d_in[19];

    const int N = in_sizes[0] / 256;        // 50000
    const int E = in_sizes[1] / 2;          // 1000000
    const int* srcIdx = edge;
    const int* dstIdx = edge + E;

    // workspace carve-out (256B aligned)
    char* ws = (char*)d_ws;
    size_t off = 0;
    auto carve = [&](size_t bytes) -> void* {
        void* p = ws + off;
        off = (off + bytes + 255) & ~(size_t)255;
        return p;
    };
    unsigned short* xb     = (unsigned short*)carve((size_t)N * 256 * 2);
    unsigned short* hb1    = (unsigned short*)carve((size_t)N * 128 * 2);
    unsigned short* hb2    = (unsigned short*)carve((size_t)N * 128 * 2);
    unsigned short* wcat1T = (unsigned short*)carve((size_t)512 * 256 * 2);
    unsigned short* wcat2T = (unsigned short*)carve((size_t)512 * 128 * 2);
    unsigned short* wlbT   = (unsigned short*)carve((size_t)64 * 128 * 2);
    float*        bcat1    = (float*)carve(512 * 4);
    float*        bcat2    = (float*)carve(512 * 4);
    float*        qkvs     = (float*)carve((size_t)N * 512 * 4);
    float*        alpha    = (float*)carve((size_t)E * 4 * 4);
    unsigned int* mkey     = (unsigned int*)carve((size_t)N * 4 * 4);
    float*        denom    = (float*)carve((size_t)N * 4 * 4);
    float*        outacc   = (float*)carve((size_t)N * 128 * 4);

    const int BT = 256;
    auto blocks = [&](long n) { return (unsigned)((n + BT - 1) / BT); };

    // ---- convert inputs / weights to bf16, fuse + transpose q|k|v|s ----
    gt_cvt_bf16<<<blocks((long)N * 256), BT, 0, stream>>>(x, xb, (long)N * 256);
    gt_concat4_wT<<<blocks(512 * 256), BT, 0, stream>>>(Wq1, Wk1, Wv1, Ws1, wcat1T, 256, 8);
    gt_concat4_wT<<<blocks(512 * 128), BT, 0, stream>>>(Wq2, Wk2, Wv2, Ws2, wcat2T, 128, 7);
    gt_concat4_b<<<1, 512, 0, stream>>>(bq1, bk1, bv1, bs1, bcat1);
    gt_concat4_b<<<1, 512, 0, stream>>>(bq2, bk2, bv2, bs2, bcat2);
    gt_cvt_wlT<<<blocks(64 * 128), BT, 0, stream>>>(Wl, wlbT);

    dim3 gemmBlk(256);
    unsigned mTiles = (unsigned)((N + TILE_M - 1) / TILE_M);

    // ================= layer 1 =================
    gt_gemm_bf16<<<dim3(mTiles, 512 / TILE_N), gemmBlk, 0, stream>>>(
        xb, 256, wcat1T, bcat1, qkvs, 512, N, 256, 512);

    gt_fill_u32<<<blocks((long)N * 4), BT, 0, stream>>>(mkey, 0u, (long)N * 4);
    gt_fill_u32<<<blocks((long)N * 4), BT, 0, stream>>>((unsigned int*)denom, 0u, (long)N * 4);
    gt_fill_u32<<<blocks((long)N * 128), BT, 0, stream>>>((unsigned int*)outacc, 0u, (long)N * 128);

    gt_edge_logits<<<blocks((long)E * 4), BT, 0, stream>>>(qkvs, srcIdx, dstIdx, alpha, mkey, E);
    gt_edge_scatter<<<blocks((long)E * 32), BT, 0, stream>>>(qkvs, alpha, mkey, srcIdx, dstIdx,
                                                             denom, outacc, E);
    gt_node_epilogue<<<blocks((long)N * 128), BT, 0, stream>>>(qkvs, outacc, denom, hb1, N);

    // ================= layer 2 =================
    gt_gemm_bf16<<<dim3(mTiles, 512 / TILE_N), gemmBlk, 0, stream>>>(
        hb1, 128, wcat2T, bcat2, qkvs, 512, N, 128, 512);

    gt_fill_u32<<<blocks((long)N * 4), BT, 0, stream>>>(mkey, 0u, (long)N * 4);
    gt_fill_u32<<<blocks((long)N * 4), BT, 0, stream>>>((unsigned int*)denom, 0u, (long)N * 4);
    gt_fill_u32<<<blocks((long)N * 128), BT, 0, stream>>>((unsigned int*)outacc, 0u, (long)N * 128);

    gt_edge_logits<<<blocks((long)E * 4), BT, 0, stream>>>(qkvs, srcIdx, dstIdx, alpha, mkey, E);
    gt_edge_scatter<<<blocks((long)E * 32), BT, 0, stream>>>(qkvs, alpha, mkey, srcIdx, dstIdx,
                                                             denom, outacc, E);
    gt_node_epilogue<<<blocks((long)N * 128), BT, 0, stream>>>(qkvs, outacc, denom, hb2, N);

    // ================= classifier head =================
    gt_gemm_bf16<<<dim3(mTiles, 1), gemmBlk, 0, stream>>>(
        hb2, 128, wlbT, bl, (float*)d_out, 16, N, 128, 16);
}
